// MambaBlock_64209761075744
// MI455X (gfx1250) — compile-verified, matrix operand order checked by reading, not run on hardware
//
#include <hip/hip_runtime.h>

#define NB    4
#define LSEQ  2048
#define HID   1024
#define DI    2048
#define DS    16
#define NTOK  (NB*LSEQ)      // 8192
#define NPROJ 33             // 1 + 2*DS

typedef __attribute__((ext_vector_type(16))) __bf16 v16bf;
typedef __attribute__((ext_vector_type(8)))  __bf16 v8bf;
typedef __attribute__((ext_vector_type(8)))  float  v8f;
typedef __attribute__((ext_vector_type(4)))  float  v4f;
typedef int v4i_g __attribute__((vector_size(16)));   // matches builtin param type

#if __has_builtin(__builtin_amdgcn_global_load_async_to_lds_b128)
#define HAVE_ASYNC 1
#else
#define HAVE_ASYNC 0
#endif

__device__ __forceinline__ void cp16_g2l(const __bf16* g, __bf16* l){
#if HAVE_ASYNC
  __builtin_amdgcn_global_load_async_to_lds_b128((v4i_g*)g, (v4i_g*)l, 0, 0);
#else
  *(v8bf*)l = *(const v8bf*)g;
#endif
}

__device__ __forceinline__ void async_fence(){
#if HAVE_ASYNC
#if __has_builtin(__builtin_amdgcn_s_wait_asynccnt)
  __builtin_amdgcn_s_wait_asynccnt(0);
#else
  asm volatile("s_wait_asynccnt 0x0" ::: "memory");
#endif
#endif
}

__device__ __forceinline__ float sigmoidf_(float x){ return 1.0f/(1.0f+__expf(-x)); }
__device__ __forceinline__ float siluf_(float x){ return x*sigmoidf_(x); }
__device__ __forceinline__ float softplusf_(float x){ return (x>20.0f)?x:log1pf(__expf(x)); }

__device__ __forceinline__ float wred(float v){
  #pragma unroll
  for (int off = 16; off > 0; off >>= 1) v += __shfl_xor(v, off, 32);
  return v;
}

// ---------------- split fp32 -> bf16 hi/lo (for weights) ----------------
__global__ void k_split(const float* __restrict__ in, __bf16* __restrict__ hi,
                        __bf16* __restrict__ lo, int n){
  int i = (blockIdx.x * blockDim.x + threadIdx.x) * 4;
  if (i >= n) return;
  v4f t = *(const v4f*)(in + i);
  #pragma unroll
  for (int e = 0; e < 4; e++){
    float fv = t[e];
    __bf16 h = (__bf16)fv;
    hi[i+e] = h;
    lo[i+e] = (__bf16)(fv - (float)h);
  }
}

// ---------------- LayerNorm 1 -> bf16 hi/lo, one wave per token ----------------
__global__ void k_ln1(const float* __restrict__ x, const float* __restrict__ g,
                      const float* __restrict__ b,
                      __bf16* __restrict__ xnh, __bf16* __restrict__ xnl){
  int wave = threadIdx.x >> 5, lane = threadIdx.x & 31;
  int tok  = blockIdx.x * 8 + wave;
  const float* xr = x + (size_t)tok * HID;
  float v[HID/32];
  float s = 0.f, s2 = 0.f;
  #pragma unroll
  for (int i = 0; i < HID/32; i++){
    v[i] = xr[lane + 32*i];
    s += v[i]; s2 += v[i]*v[i];
  }
  s = wred(s); s2 = wred(s2);
  float mu = s * (1.0f/HID);
  float var = s2 * (1.0f/HID) - mu*mu;
  float rs = rsqrtf(var + 1e-5f);
  #pragma unroll
  for (int i = 0; i < HID/32; i++){
    int h = lane + 32*i;
    float nv = (v[i] - mu) * rs * g[h] + b[h];
    __bf16 hb = (__bf16)nv;
    xnh[(size_t)tok*HID + h] = hb;
    xnl[(size_t)tok*HID + h] = (__bf16)(nv - (float)hb);
  }
}

// ---------------- bf16x3 WMMA GEMM, async double-buffered LDS ----------------
// C[M x N] = X[M x K] * W[N x K]^T ; X/W pre-split into bf16 hi/lo.
// Block: 256 thr = 8 waves, tile 128(M) x 64(N), K-chunk 32, double buffered.
#define BM   128
#define BN   64
#define KC   32
#define KSTR 40   // padded row stride (bf16) -> 80B, keeps 16B alignment

__device__ __forceinline__ v16bf frag_ld(const __bf16* base, int row, int lane){
  // 16-bit A/B layout: lanes 0-15 hold K 0..7 & 16..23, lanes 16-31 hold K 8..15 & 24..31
  const __bf16* p = base + row*KSTR + ((lane & 16) >> 1);
  v8bf a = *(const v8bf*)p;
  v8bf c = *(const v8bf*)(p + 16);
  return __builtin_shufflevector(a, c, 0,1,2,3,4,5,6,7,8,9,10,11,12,13,14,15);
}

__global__ void __launch_bounds__(256)
k_gemm(const __bf16* __restrict__ Xh, const __bf16* __restrict__ Xl,
       const __bf16* __restrict__ Wh, const __bf16* __restrict__ Wl,
       int K, int N, int mode,
       float* __restrict__ out0, float* __restrict__ out1,
       const float* __restrict__ resid)
{
  __shared__ __align__(16) __bf16 sA[2][2][BM*KSTR];  // [buf][hi/lo]
  __shared__ __align__(16) __bf16 sB[2][2][BN*KSTR];

  int tid = threadIdx.x, lane = tid & 31, w = tid >> 5;
  int wm = w & 1;     // 2 M-subtiles of 64
  int wn = w >> 1;    // 4 N-subtiles of 16
  int m_blk = blockIdx.x * BM, n_blk = blockIdx.y * BN;

  auto stage = [&](int buf, int k0){
    // A tile: 128 rows x 4 chunks of 8 bf16 -> 512 chunks, 2 per thread
    #pragma unroll
    for (int q = 0; q < 2; q++){
      int cid = tid + 256*q;
      int row = cid >> 2, c = (cid & 3) * 8;
      size_t go = (size_t)(m_blk + row) * K + k0 + c;
      cp16_g2l(Xh + go, &sA[buf][0][row*KSTR + c]);
      cp16_g2l(Xl + go, &sA[buf][1][row*KSTR + c]);
    }
    // B tile: 64 rows x 4 chunks -> 256 chunks, 1 per thread
    {
      int row = tid >> 2, c = (tid & 3) * 8;
      size_t go = (size_t)(n_blk + row) * K + k0 + c;
      cp16_g2l(Wh + go, &sB[buf][0][row*KSTR + c]);
      cp16_g2l(Wl + go, &sB[buf][1][row*KSTR + c]);
    }
  };

  v8f acc[4];
  #pragma unroll
  for (int mt = 0; mt < 4; mt++)
    acc[mt] = (v8f){0.f,0.f,0.f,0.f,0.f,0.f,0.f,0.f};

  int nk = K / KC;
  stage(0, 0);
  async_fence();
  __syncthreads();

  for (int ki = 0; ki < nk; ki++){
    int cur = ki & 1;
    if (ki + 1 < nk) stage(cur ^ 1, (ki + 1) * KC);

    const __bf16* Ah = sA[cur][0];
    const __bf16* Al = sA[cur][1];
    const __bf16* Bh = sB[cur][0];
    const __bf16* Bl = sB[cur][1];

    int c = wn*16 + (lane & 15);
    v16bf bh = frag_ld(Bh, c, lane);
    v16bf bl = frag_ld(Bl, c, lane);

    v16bf ah[4], al[4];
    #pragma unroll
    for (int mt = 0; mt < 4; mt++){
      int r = wm*64 + mt*16 + (lane & 15);
      ah[mt] = frag_ld(Ah, r, lane);
      al[mt] = frag_ld(Al, r, lane);
    }
    #pragma unroll
    for (int mt = 0; mt < 4; mt++){
      acc[mt] = __builtin_amdgcn_wmma_f32_16x16x32_bf16(
          false, ah[mt], false, bh, (short)0, acc[mt], false, false);
      acc[mt] = __builtin_amdgcn_wmma_f32_16x16x32_bf16(
          false, ah[mt], false, bl, (short)0, acc[mt], false, false);
      acc[mt] = __builtin_amdgcn_wmma_f32_16x16x32_bf16(
          false, al[mt], false, bh, (short)0, acc[mt], false, false);
    }

    async_fence();
    __syncthreads();
  }

  // epilogue: VGPR r holds M=r (lanes 0-15) / M=r+8 (lanes 16-31); N = lane&15
  int rofs = (lane & 16) >> 1;   // 0 or 8
  int col  = n_blk + wn*16 + (lane & 15);
  #pragma unroll
  for (int mt = 0; mt < 4; mt++){
    #pragma unroll
    for (int r = 0; r < 8; r++){
      int row = m_blk + wm*64 + mt*16 + rofs + r;
      float v = acc[mt][r];
      if (mode == 0){
        if (col < DI) out0[(size_t)row*DI + col] = siluf_(v);
        else          out1[(size_t)row*DI + (col - DI)] = v;
      } else {
        out0[(size_t)row*N + col] = v + resid[(size_t)row*N + col];
      }
    }
  }
}

// ---------------- fused double causal depthwise conv + SiLU ----------------
__global__ void k_conv(const float* __restrict__ xs,
                       const float* __restrict__ w0, const float* __restrict__ b0,
                       const float* __restrict__ w1, const float* __restrict__ b1,
                       float* __restrict__ xc){
  int gid = blockIdx.x * blockDim.x + threadIdx.x;   // 0..NB*DI-1
  int b = gid / DI, d = gid % DI;
  float c0[4], c1[4];
  #pragma unroll
  for (int j = 0; j < 4; j++){ c0[j] = w0[d*4 + j]; c1[j] = w1[d*4 + j]; }
  float bb0 = b0[d], bb1 = b1[d];
  float x1 = 0.f, x2 = 0.f, x3 = 0.f;
  float y1 = 0.f, y2 = 0.f, y3 = 0.f;
  const float* xin = xs + (size_t)b * LSEQ * DI + d;
  float*       xo  = xc + (size_t)b * LSEQ * DI + d;
  for (int l = 0; l < LSEQ; l++){
    float xt = xin[(size_t)l * DI];
    float u = siluf_(c0[0]*x3 + c0[1]*x2 + c0[2]*x1 + c0[3]*xt + bb0);
    float v = siluf_(c1[0]*y3 + c1[1]*y2 + c1[2]*y1 + c1[3]*u  + bb1);
    xo[(size_t)l * DI] = v;
    x3 = x2; x2 = x1; x1 = xt;
    y3 = y2; y2 = y1; y1 = u;
  }
}

// ---------------- x_proj: 33 outputs per token, one wave per token ----------------
__global__ void k_xproj(const float* __restrict__ xc, const float* __restrict__ xw,
                        float* __restrict__ xp){
  int wave = threadIdx.x >> 5, lane = threadIdx.x & 31;
  int tok  = blockIdx.x * 8 + wave;
  const float* xr = xc + (size_t)tok * DI;
  float acc[NPROJ];
  #pragma unroll
  for (int o = 0; o < NPROJ; o++) acc[o] = 0.f;
  for (int i = 0; i < DI/32; i++){
    int d = lane + 32*i;
    float xv = xr[d];
    #pragma unroll
    for (int o = 0; o < NPROJ; o++) acc[o] += xv * xw[o*DI + d];
  }
  #pragma unroll
  for (int o = 0; o < NPROJ; o++){
    float s = wred(acc[o]);
    if (lane == 0) xp[(size_t)tok * NPROJ + o] = s;
  }
}

// ---------------- selective scan: one lane per (b,d) channel ----------------
__global__ void k_scan(const float* __restrict__ xc, const float* __restrict__ xp,
                       const float* __restrict__ dtw, const float* __restrict__ dtb,
                       const float* __restrict__ A_log, const float* __restrict__ Dp,
                       float* __restrict__ yD){
  int gid = blockIdx.x * blockDim.x + threadIdx.x;
  int b = gid / DI, d = gid % DI;
  float A[DS], h[DS];
  #pragma unroll
  for (int n = 0; n < DS; n++){ A[n] = -__expf(A_log[d*DS + n]); h[n] = 0.f; }
  float wdt = dtw[d], bdt = dtb[d], Dd = Dp[d];
  const float* xin = xc + (size_t)b * LSEQ * DI + d;
  const float* pr  = xp + (size_t)b * LSEQ * NPROJ;
  float*       yo  = yD + (size_t)b * LSEQ * DI + d;
  for (int l = 0; l < LSEQ; l++){
    const float* p = pr + l * NPROJ;
    float dt = softplusf_(p[0]*wdt + bdt);
    float xt = xin[(size_t)l * DI];
    float y  = 0.f;
    #pragma unroll
    for (int n = 0; n < DS; n++){
      h[n] += dt * (h[n]*A[n] + xt*p[1 + n]);
      y    += h[n] * p[17 + n];
    }
    yo[(size_t)l * DI] = y + xt*Dd;
  }
}

// ---------------- LayerNorm 2 + sigmoid gate -> bf16 hi/lo ----------------
__global__ void k_ln2gate(const float* __restrict__ yD, const float* __restrict__ res,
                          const float* __restrict__ g, const float* __restrict__ bb,
                          __bf16* __restrict__ yh, __bf16* __restrict__ yl){
  int wave = threadIdx.x >> 5, lane = threadIdx.x & 31;
  int tok  = blockIdx.x * 8 + wave;
  const float* yr = yD + (size_t)tok * DI;
  const float* rr = res + (size_t)tok * DI;
  float s = 0.f, s2 = 0.f;
  for (int i = 0; i < DI/32; i++){
    float v = yr[lane + 32*i];
    s += v; s2 += v*v;
  }
  s = wred(s); s2 = wred(s2);
  float mu = s * (1.0f/DI);
  float var = s2 * (1.0f/DI) - mu*mu;
  float rs = rsqrtf(var + 1e-5f);
  for (int i = 0; i < DI/32; i++){
    int dd = lane + 32*i;
    float v  = yr[dd];
    float nv = ((v - mu) * rs * g[dd] + bb[dd]) * sigmoidf_(rr[dd]);
    __bf16 hb = (__bf16)nv;
    yh[(size_t)tok*DI + dd] = hb;
    yl[(size_t)tok*DI + dd] = (__bf16)(nv - (float)hb);
  }
}

extern "C" void kernel_launch(void* const* d_in, const int* in_sizes, int n_in,
                              void* d_out, int out_size, void* d_ws, size_t ws_size,
                              hipStream_t stream){
  (void)in_sizes; (void)n_in; (void)out_size; (void)ws_size;
  const float* x    = (const float*)d_in[0];
  const float* n1g  = (const float*)d_in[1];
  const float* n1b  = (const float*)d_in[2];
  const float* win  = (const float*)d_in[3];
  const float* cw0  = (const float*)d_in[4];
  const float* cb0  = (const float*)d_in[5];
  const float* cw1  = (const float*)d_in[6];
  const float* cb1  = (const float*)d_in[7];
  const float* xw   = (const float*)d_in[8];
  const float* dtw  = (const float*)d_in[9];
  const float* dtb  = (const float*)d_in[10];
  const float* Alog = (const float*)d_in[11];
  const float* Dp   = (const float*)d_in[12];
  const float* n2g  = (const float*)d_in[13];
  const float* n2b  = (const float*)d_in[14];
  const float* wout = (const float*)d_in[15];
  float* out = (float*)d_out;

  // workspace layout (bytes), ~266 MB total:
  //   0      : xs (silu x_inner, later reused as yD)   64MB fp32
  //   64MB   : res                                     64MB fp32
  //   128MB  : xc                                      64MB fp32
  //   192MB  : xp                                      ~1.1MB fp32
  //   194MB  : region A (phase1: xnh|xnl|winh|winl ; phase2: y2h|y2l)  64MB
  //   258MB  : wouth (4MB) | woutl (4MB)
  char* wsb = (char*)d_ws;
  float* xs  = (float*)wsb;
  float* res = (float*)(wsb + ((size_t)64  << 20));
  float* xc  = (float*)(wsb + ((size_t)128 << 20));
  float* xp  = (float*)(wsb + ((size_t)192 << 20));
  char*  rgA = wsb + ((size_t)194 << 20);
  __bf16* xnh  = (__bf16*)rgA;                         // 16MB
  __bf16* xnl  = (__bf16*)(rgA + ((size_t)16 << 20));  // 16MB
  __bf16* winh = (__bf16*)(rgA + ((size_t)32 << 20));  // 8MB
  __bf16* winl = (__bf16*)(rgA + ((size_t)40 << 20));  // 8MB
  __bf16* y2h  = (__bf16*)rgA;                         // 32MB (after GEMM1)
  __bf16* y2l  = (__bf16*)(rgA + ((size_t)32 << 20));  // 32MB
  __bf16* wouth= (__bf16*)(wsb + ((size_t)258 << 20)); // 4MB
  __bf16* woutl= (__bf16*)(wsb + ((size_t)262 << 20)); // 4MB

  k_split<<<(2*DI*HID)/1024, 256, 0, stream>>>(win,  winh,  winl,  2*DI*HID);
  k_split<<<(HID*DI)/1024,   256, 0, stream>>>(wout, wouth, woutl, HID*DI);
  k_ln1<<<NTOK/8, 256, 0, stream>>>(x, n1g, n1b, xnh, xnl);

  k_gemm<<<dim3(NTOK/BM, (2*DI)/BN), 256, 0, stream>>>(
      xnh, xnl, winh, winl, HID, 2*DI, 0, xs, res, nullptr);

  k_conv<<<(NB*DI)/256, 256, 0, stream>>>(xs, cw0, cb0, cw1, cb1, xc);
  k_xproj<<<NTOK/8, 256, 0, stream>>>(xc, xw, xp);
  k_scan<<<(NB*DI)/256, 256, 0, stream>>>(xc, xp, dtw, dtb, Alog, Dp, xs /*yD*/);
  k_ln2gate<<<NTOK/8, 256, 0, stream>>>(xs, res, n2g, n2b, y2h, y2l);

  k_gemm<<<dim3(NTOK/BM, HID/BN), 256, 0, stream>>>(
      y2h, y2l, wouth, woutl, DI, HID, 1, out, nullptr, x);
}
